// SpikingLayer_36764920053771
// MI455X (gfx1250) — matrix operand order repaired and checked by reading, not compiled, not run on hardware
//
#include <hip/hip_runtime.h>

#define B_  64
#define S_  512
#define D_  1024
#define H_  1024
#define G4  4096      /* 4*H */
#define KC  2048      /* D + H */
#define LDT 72        /* padded LDS row stride (halves): 64 data + 8 pad = 144B */
#define LDG 65        /* padded LDS row stride (floats) for gate staging */

typedef __attribute__((ext_vector_type(16))) __bf16 bf16x16;
typedef __attribute__((ext_vector_type(8)))  float  f32x8;
typedef __attribute__((ext_vector_type(4)))  unsigned u32x4;
typedef __attribute__((ext_vector_type(8)))  int  i32x8;
typedef __attribute__((ext_vector_type(4)))  int  i32x4;

__device__ __forceinline__ unsigned short f2bf(float f) {
  union { float f; unsigned u; } v; v.f = f;
  unsigned u = v.u;
  unsigned r = u + 0x7FFFu + ((u >> 16) & 1u);   // RNE
  return (unsigned short)(r >> 16);
}

// ---- CDNA5 async global->LDS (per-lane 16B), tracked by ASYNCcnt ----
__device__ __forceinline__ void async_ld_b128(const unsigned short* lds_dst,
                                              const unsigned short* gsrc) {
  unsigned loff = (unsigned)(unsigned long long)(const void*)lds_dst; // LDS offset
  unsigned long long ga = (unsigned long long)(const void*)gsrc;
  asm volatile("global_load_async_to_lds_b128 %0, %1, off"
               :: "v"(loff), "v"(ga) : "memory");
}
__device__ __forceinline__ void wait_async0() {
  asm volatile("s_wait_asynccnt 0x0" ::: "memory");
}

// ---- Tensor Data Mover: 3D tile (4 gate blocks x 16 rows x 64 k bf16) ----
// from Wcat[4096][2048] into LDS with 4-DWORD pad every 32 DWORDs (row stride
// 144B == LDT halves). Issued by one wave; tracked by TENSORcnt.
__device__ __forceinline__ void tdm_load_wtile(const unsigned short* gsrc,
                                               const unsigned short* lds_dst) {
  unsigned long long ga = (unsigned long long)(const void*)gsrc;
  unsigned loff = (unsigned)(unsigned long long)(const void*)lds_dst;
  u32x4 g0;
  g0[0] = 1u;                                              // count=1 (user D#)
  g0[1] = loff;                                            // lds_addr
  g0[2] = (unsigned)(ga & 0xFFFFFFFFull);                  // global_addr[31:0]
  g0[3] = (unsigned)((ga >> 32) & 0x01FFFFFFull) | (2u << 30); // [56:32], type=2
  i32x8 g1;
  g1[0] = (int)((1u << 16) |        // data_size = 2 bytes
                (1u << 20) |        // pad_enable
                (4u << 22) |        // pad_interval: every 32 DWORDs (one row)
                (3u << 25));        // pad_amount: 4 DWORDs (16B)
  g1[1] = (int)((2048u & 0xFFFFu) << 16);  // tensor_dim0 = 2048 (lo16)
  g1[2] = (int)((1024u & 0xFFFFu) << 16);  // dim0 hi16=0 ; tensor_dim1 = 1024 (lo16)
  g1[3] = (int)(64u << 16);                // dim1 hi16=0 ; tile_dim0 = 64
  g1[4] = (int)(16u | (4u << 16));         // tile_dim1 = 16 ; tile_dim2 = 4
  g1[5] = (int)2048;                       // tensor_dim0_stride lo32 = 2048
  g1[6] = (int)((2048u & 0xFFFFu) << 16);  // d0s hi16=0 ; tensor_dim1_stride lo16
  g1[7] = 0;                               // tensor_dim1_stride hi32 = 0
  i32x4 g2;
  g2[0] = 4;                               // tensor_dim2 = 4 (gate blocks)
  g2[1] = 0;                               // tensor_dim3 = 0
  g2[2] = (int)(1024u * 2048u);            // tensor_dim2_stride = H*KC elements
  g2[3] = 0;                               // d2s hi ; tile_dim3 = 0
  i32x4 g3 = {0, 0, 0, 0};
#if defined(__clang_major__) && (__clang_major__ >= 23)
  i32x8 g4 = {0, 0, 0, 0, 0, 0, 0, 0};
  __builtin_amdgcn_tensor_load_to_lds(g0, g1, g2, g3, g4, 0);
#else
  __builtin_amdgcn_tensor_load_to_lds(g0, g1, g2, g3, 0);
#endif
}

// A fragment (16x32 bf16): lane half h: elems e -> K=16h+e
__device__ __forceinline__ bf16x16 ld_frag_a(const unsigned short* p0, int lane) {
  int m = lane & 15, h = lane >> 4;
  const unsigned short* p = p0 + m * LDT;
  union { unsigned short s[16]; bf16x16 v; } u;
#pragma unroll
  for (int e = 0; e < 8; ++e) {
    u.s[e]     = p[16 * h + e];
    u.s[8 + e] = p[16 * h + 8 + e];
  }
  return u.v;
}

// B fragment (32x16 bf16), LDS tile stored [N][K]: lane n, elems e -> K=16h+e
__device__ __forceinline__ bf16x16 ld_frag_b(const unsigned short* p0, int lane) {
  int n = lane & 15, h = lane >> 4;
  const unsigned short* p = p0 + n * LDT + 16 * h;
  union { unsigned short s[16]; bf16x16 v; } u;
#pragma unroll
  for (int e = 0; e < 16; ++e) u.s[e] = p[e];
  return u.v;
}

__device__ __forceinline__ f32x8 wmma_bf16(bf16x16 a, bf16x16 b, f32x8 c) {
  return __builtin_amdgcn_wmma_f32_16x16x32_bf16(false, a, false, b, (short)0, c,
                                                 false, false);
}

__device__ __forceinline__ float sigmoidf_(float x) { return 1.0f / (1.0f + __expf(-x)); }

// ---------------- conversion / setup kernels ----------------

__global__ void k_cvt_bf16(const float* __restrict__ src,
                           unsigned short* __restrict__ dst, long n) {
  long i = (long)blockIdx.x * blockDim.x + threadIdx.x;
  long stride = (long)gridDim.x * blockDim.x;
  for (; i < n; i += stride) dst[i] = f2bf(src[i]);
}

__global__ void k_build_wcat(const float* __restrict__ Wih,
                             const float* __restrict__ Whh,
                             unsigned short* __restrict__ wcat) {
  long n = (long)G4 * KC;
  long i = (long)blockIdx.x * blockDim.x + threadIdx.x;
  long stride = (long)gridDim.x * blockDim.x;
  for (; i < n; i += stride) {
    int g = (int)(i >> 11), k = (int)(i & 2047);
    float v = (k < D_) ? Wih[(long)g * D_ + k] : Whh[(long)g * H_ + (k - D_)];
    wcat[i] = f2bf(v);
  }
}

__global__ void k_bias(const float* __restrict__ bih, const float* __restrict__ bhh,
                       float* __restrict__ biasg) {
  int i = blockIdx.x * blockDim.x + threadIdx.x;
  if (i < G4) biasg[i] = bih[i] + bhh[i];
}

__global__ void k_init_state(const float* __restrict__ syn_in,
                             const float* __restrict__ mem_in,
                             float* __restrict__ synws,
                             unsigned short* __restrict__ mem0,
                             unsigned* __restrict__ bar) {
  int i = blockIdx.x * blockDim.x + threadIdx.x;
  if (i < B_ * H_) { synws[i] = syn_in[i]; mem0[i] = f2bf(mem_in[i]); }
  if (i == 0) { bar[0] = 0u; bar[1] = 0u; }
}

// ---------------- persistent recurrence kernel ----------------
// Grid: 64 blocks x 256 threads. Block b owns h-columns [16b, 16b+16).
// gates[64, 4x16] = [x_t ; mem] (K=2048) x Wcat^T via bf16 WMMA.
// W tile via TDM; A tile via async global->LDS.

__global__ void __launch_bounds__(256) k_recur(
    const unsigned short* __restrict__ xb,     // [B*S*D] bf16
    const unsigned short* __restrict__ wcat,   // [4096][2048] bf16
    const float* __restrict__ biasg,           // [4096] b_ih+b_hh
    const float* __restrict__ thr_p,
    unsigned short* __restrict__ membuf,       // [2][B*H] bf16 ping-pong
    float* __restrict__ synws,                 // [B*H]
    unsigned short* __restrict__ spks,         // [B*S*H] bf16
    float* __restrict__ out_syn,
    float* __restrict__ out_mem,
    unsigned* __restrict__ bar) {
  __shared__ alignas(16) unsigned short lds_a[64 * LDT];
  __shared__ alignas(16) unsigned short lds_w[64 * LDT];
  __shared__ float lds_g[64 * LDG];

  const int tid = threadIdx.x, lane = tid & 31, wave = tid >> 5;
  const int mt = wave >> 1, ntB = (wave & 1) * 2;
  const int h0 = blockIdx.x * 16;
  const float thr = thr_p[0];
  const int nwg = gridDim.x;

  for (int t = 0; t < S_; ++t) {
    const unsigned short* mb = membuf + (t & 1) * (B_ * H_);
    unsigned short* mbn = membuf + ((t + 1) & 1) * (B_ * H_);
    f32x8 acc0 = {0.f,0.f,0.f,0.f,0.f,0.f,0.f,0.f};
    f32x8 acc1 = {0.f,0.f,0.f,0.f,0.f,0.f,0.f,0.f};

    for (int kc = 0; kc < KC; kc += 64) {
      // A tile (64 rows x 64 k) via async global->LDS
#pragma unroll
      for (int q = 0; q < 2; ++q) {
        int c = tid + q * 256;
        int row = c >> 3, kch = (c & 7) * 8;
        const unsigned short* asrc = (kc < D_)
            ? xb + ((long)row * S_ + t) * D_ + kc + kch
            : mb + row * H_ + (kc - D_) + kch;
        async_ld_b128(&lds_a[row * LDT + kch], asrc);
      }
      // W tile (4 gate blocks x 16 rows x 64 k) via Tensor Data Mover (wave 0)
      if (wave == 0) {
        tdm_load_wtile(wcat + ((long)h0 * KC + kc), lds_w);
      }
      wait_async0();
      if (wave == 0) __builtin_amdgcn_s_wait_tensorcnt(0);
      __syncthreads();
#pragma unroll
      for (int kk = 0; kk < 64; kk += 32) {
        bf16x16 a = ld_frag_a(&lds_a[mt * 16 * LDT + kk], lane);
        acc0 = wmma_bf16(a, ld_frag_b(&lds_w[ntB * 16 * LDT + kk], lane), acc0);
        acc1 = wmma_bf16(a, ld_frag_b(&lds_w[(ntB + 1) * 16 * LDT + kk], lane), acc1);
      }
      __syncthreads();
    }

    // stage gates (f32) to LDS: rows = batch b, cols = 4 gate blocks x 16
    {
      int col = lane & 15, half = lane >> 4;
#pragma unroll
      for (int r = 0; r < 8; ++r) {
        int m = mt * 16 + r + 8 * half;
        lds_g[m * LDG + ntB * 16 + col]       = acc0[r];
        lds_g[m * LDG + (ntB + 1) * 16 + col] = acc1[r];
      }
    }
    __syncthreads();

    // elementwise SLSTM update for this block's 16 h-columns x 64 batches
#pragma unroll
    for (int q = 0; q < 4; ++q) {
      int idx = tid + q * 256;                 // 0..1023
      int b = idx >> 4, hl = idx & 15, h = h0 + hl;
      float gi = lds_g[b * LDG + hl]      + biasg[h];
      float gf = lds_g[b * LDG + 16 + hl] + biasg[H_ + h];
      float gg = lds_g[b * LDG + 32 + hl] + biasg[2 * H_ + h];
      float go = lds_g[b * LDG + 48 + hl] + biasg[3 * H_ + h];
      float syn_n = sigmoidf_(gf) * synws[b * H_ + h] + sigmoidf_(gi) * tanhf(gg);
      float mem_n = sigmoidf_(go) * tanhf(syn_n);
      synws[b * H_ + h] = syn_n;
      mbn[b * H_ + h] = f2bf(mem_n);
      spks[((long)b * S_ + t) * H_ + h] =
          (mem_n - thr) > 0.f ? (unsigned short)0x3F80u : (unsigned short)0u;
      if (t == S_ - 1) { out_syn[b * H_ + h] = syn_n; out_mem[b * H_ + h] = mem_n; }
    }
    __syncthreads();
    __threadfence();

    // device-scope barrier (all 64 persistent workgroups)
    if (tid == 0) {
      unsigned arrived = atomicAdd(&bar[0], 1u);
      if (arrived == (unsigned)(nwg - 1)) {
        bar[0] = 0u;
        __threadfence();
        atomicAdd(&bar[1], 1u);
      } else {
        while (__atomic_load_n(&bar[1], __ATOMIC_RELAXED) < (unsigned)(t + 1)) {
          __builtin_amdgcn_s_sleep(1);
        }
      }
      __threadfence();
    }
    __syncthreads();
  }
}

// -------- h = spks @ lin_W^T + lin_b (bf16 out) --------
// 128x64 block tile, 4 WMMA tiles per wave, double-buffered async LDS staging.

__global__ void __launch_bounds__(256) k_gemm_lin(
    const unsigned short* __restrict__ A,      // [M][1024] bf16 (spks)
    const unsigned short* __restrict__ W,      // [1024][1024] bf16
    const float* __restrict__ bias,
    unsigned short* __restrict__ Hout) {
  __shared__ alignas(16) unsigned short lds_a[2][128 * LDT];
  __shared__ alignas(16) unsigned short lds_w[2][64 * LDT];
  const int tid = threadIdx.x, lane = tid & 31, wave = tid >> 5;
  const long m0 = (long)blockIdx.x * 128;
  const int n0 = blockIdx.y * 64;

  auto issue = [&](int buf, int kc) {
#pragma unroll
    for (int q = 0; q < 4; ++q) {             // A: 128 rows x 64 k
      int c = tid + q * 256;
      int row = c >> 3, kch = (c & 7) * 8;
      async_ld_b128(&lds_a[buf][row * LDT + kch],
                    A + (m0 + row) * H_ + kc + kch);
    }
#pragma unroll
    for (int q = 0; q < 2; ++q) {             // W: 64 rows x 64 k
      int c = tid + q * 256;
      int row = c >> 3, kch = (c & 7) * 8;
      async_ld_b128(&lds_w[buf][row * LDT + kch],
                    W + (long)(n0 + row) * H_ + kc + kch);
    }
  };

  f32x8 acc[4];
#pragma unroll
  for (int nt = 0; nt < 4; ++nt) acc[nt] = (f32x8){0.f,0.f,0.f,0.f,0.f,0.f,0.f,0.f};

  issue(0, 0);
  wait_async0();
  __syncthreads();
  for (int kc = 0, it = 0; kc < H_; kc += 64, ++it) {
    int cur = it & 1;
    if (kc + 64 < H_) issue(cur ^ 1, kc + 64);
#pragma unroll
    for (int kk = 0; kk < 64; kk += 32) {
      bf16x16 a = ld_frag_a(&lds_a[cur][wave * 16 * LDT + kk], lane);
#pragma unroll
      for (int nt = 0; nt < 4; ++nt)
        acc[nt] = wmma_bf16(a, ld_frag_b(&lds_w[cur][nt * 16 * LDT + kk], lane),
                            acc[nt]);
    }
    wait_async0();
    __syncthreads();
  }
  int col = lane & 15, half = lane >> 4;
#pragma unroll
  for (int nt = 0; nt < 4; ++nt) {
    int n = n0 + nt * 16 + col;
    float bn = bias[n];
#pragma unroll
    for (int r = 0; r < 8; ++r) {
      long m = m0 + wave * 16 + r + 8 * half;
      Hout[m * H_ + n] = f2bf(acc[nt][r] + bn);
    }
  }
}

// ------ out = DyT( [h | x] @ mix_W^T + mix_b )  (fp32 out) ------

__global__ void __launch_bounds__(256) k_gemm_mix(
    const unsigned short* __restrict__ Hb,     // [M][1024] bf16
    const unsigned short* __restrict__ Xb,     // [M][1024] bf16
    const unsigned short* __restrict__ W,      // [1024][2048] bf16
    const float* __restrict__ mixb,
    const float* __restrict__ alpha_p,
    const float* __restrict__ gamma,
    const float* __restrict__ beta,
    float* __restrict__ out) {
  __shared__ alignas(16) unsigned short lds_a[2][128 * LDT];
  __shared__ alignas(16) unsigned short lds_w[2][64 * LDT];
  const int tid = threadIdx.x, lane = tid & 31, wave = tid >> 5;
  const long m0 = (long)blockIdx.x * 128;
  const int n0 = blockIdx.y * 64;
  const float alpha = alpha_p[0];

  auto issue = [&](int buf, int kc) {
#pragma unroll
    for (int q = 0; q < 4; ++q) {
      int c = tid + q * 256;
      int row = c >> 3, kch = (c & 7) * 8;
      const unsigned short* asrc = (kc < H_)
          ? Hb + (m0 + row) * H_ + kc + kch
          : Xb + (m0 + row) * D_ + (kc - H_) + kch;
      async_ld_b128(&lds_a[buf][row * LDT + kch], asrc);
    }
#pragma unroll
    for (int q = 0; q < 2; ++q) {
      int c = tid + q * 256;
      int row = c >> 3, kch = (c & 7) * 8;
      async_ld_b128(&lds_w[buf][row * LDT + kch],
                    W + (long)(n0 + row) * KC + kc + kch);
    }
  };

  f32x8 acc[4];
#pragma unroll
  for (int nt = 0; nt < 4; ++nt) acc[nt] = (f32x8){0.f,0.f,0.f,0.f,0.f,0.f,0.f,0.f};

  issue(0, 0);
  wait_async0();
  __syncthreads();
  for (int kc = 0, it = 0; kc < KC; kc += 64, ++it) {
    int cur = it & 1;
    if (kc + 64 < KC) issue(cur ^ 1, kc + 64);
#pragma unroll
    for (int kk = 0; kk < 64; kk += 32) {
      bf16x16 a = ld_frag_a(&lds_a[cur][wave * 16 * LDT + kk], lane);
#pragma unroll
      for (int nt = 0; nt < 4; ++nt)
        acc[nt] = wmma_bf16(a, ld_frag_b(&lds_w[cur][nt * 16 * LDT + kk], lane),
                            acc[nt]);
    }
    wait_async0();
    __syncthreads();
  }
  int col = lane & 15, half = lane >> 4;
#pragma unroll
  for (int nt = 0; nt < 4; ++nt) {
    int n = n0 + nt * 16 + col;
    float bn = mixb[n], gn = gamma[n], be = beta[n];
#pragma unroll
    for (int r = 0; r < 8; ++r) {
      long m = m0 + wave * 16 + r + 8 * half;
      float v = acc[nt][r] + bn;
      out[m * H_ + n] = gn * tanhf(alpha * v) + be;
    }
  }
}

// ---------------- launch ----------------

extern "C" void kernel_launch(void* const* d_in, const int* in_sizes, int n_in,
                              void* d_out, int out_size, void* d_ws, size_t ws_size,
                              hipStream_t stream) {
  (void)in_sizes; (void)n_in; (void)out_size; (void)ws_size;
  const float* x     = (const float*)d_in[0];
  const float* syn0  = (const float*)d_in[1];
  const float* mem0f = (const float*)d_in[2];
  const float* Wih   = (const float*)d_in[3];
  const float* Whh   = (const float*)d_in[4];
  const float* bih   = (const float*)d_in[5];
  const float* bhh   = (const float*)d_in[6];
  const float* thr   = (const float*)d_in[7];
  const float* linW  = (const float*)d_in[8];
  const float* linb  = (const float*)d_in[9];
  const float* mixW  = (const float*)d_in[10];
  const float* mixb  = (const float*)d_in[11];
  const float* alpha = (const float*)d_in[12];
  const float* gamma = (const float*)d_in[13];
  const float* beta  = (const float*)d_in[14];
  float* out = (float*)d_out;

  char* ws = (char*)d_ws;
  size_t off = 0;
  auto alloc = [&](size_t bytes) -> void* {
    void* p = ws + off;
    off += (bytes + 255) & ~(size_t)255;
    return p;
  };
  unsigned short* xb     = (unsigned short*)alloc((size_t)B_ * S_ * D_ * 2);
  unsigned short* wcat   = (unsigned short*)alloc((size_t)G4 * KC * 2);
  float*          biasg  = (float*)alloc((size_t)G4 * 4);
  unsigned short* membuf = (unsigned short*)alloc((size_t)2 * B_ * H_ * 2);
  float*          synws  = (float*)alloc((size_t)B_ * H_ * 4);
  unsigned short* spks   = (unsigned short*)alloc((size_t)B_ * S_ * H_ * 2);
  unsigned short* hb     = (unsigned short*)alloc((size_t)B_ * S_ * H_ * 2);
  unsigned short* linWb  = (unsigned short*)alloc((size_t)H_ * H_ * 2);
  unsigned short* mixWb  = (unsigned short*)alloc((size_t)H_ * KC * 2);
  unsigned*       bar    = (unsigned*)alloc(256);

  k_cvt_bf16 <<<2048, 256, 0, stream>>>(x, xb, (long)B_ * S_ * D_);
  k_build_wcat<<<2048, 256, 0, stream>>>(Wih, Whh, wcat);
  k_bias     <<<16,   256, 0, stream>>>(bih, bhh, biasg);
  k_cvt_bf16 <<<512,  256, 0, stream>>>(linW, linWb, (long)H_ * H_);
  k_cvt_bf16 <<<1024, 256, 0, stream>>>(mixW, mixWb, (long)H_ * KC);
  k_init_state<<<256, 256, 0, stream>>>(syn0, mem0f, synws, membuf, bar);

  float* out_syn = out + (size_t)B_ * S_ * H_;
  float* out_mem = out_syn + (size_t)B_ * H_;
  k_recur<<<64, 256, 0, stream>>>(xb, wcat, biasg, thr, membuf, synws, spks,
                                  out_syn, out_mem, bar);

  k_gemm_lin<<<dim3(256, 16), 256, 0, stream>>>(spks, linWb, linb, hb);
  k_gemm_mix<<<dim3(256, 16), 256, 0, stream>>>(hb, xb, mixWb, mixb, alpha,
                                                gamma, beta, out);
}
